// GATLayer_36292473651937
// MI455X (gfx1250) — matrix operand order repaired and checked by reading, not compiled, not run on hardware
//
#include <hip/hip_runtime.h>
#include <hip/hip_bf16.h>

// GAT layer, fused single-pass over the 256MB adj matrix (the bandwidth floor),
// with the att@Wh matmul done via v_wmma_f32_16x16x32_f16 (f16 P-tiles / WhT).
// Round 3 -> 4: revert the x2 unroll (it caused register-pressure-induced
// v_nop stalls after every v_exp_f32); keep the scalar backedge; split lsum
// into 4 independent partial accumulators; phase the element computation so
// every v_exp_f32 has independent neighbors to co-issue with.

#define NN    8192
#define FIN   94
#define FOUT  16
#define ALPHA 0.2f
#define LOG2E 1.44269504088896f

typedef __attribute__((ext_vector_type(16))) _Float16 v16h;
typedef __attribute__((ext_vector_type(8)))  _Float16 v8h;
typedef __attribute__((ext_vector_type(8)))  float    v8f;

// ---------------- Kernel 1: Wh, WhT(f16), s_src, s_dst, per-block max(s_dst) ----
__global__ __launch_bounds__(256) void gat_prep(
    const float* __restrict__ h, const float* __restrict__ W,
    const float* __restrict__ attn,
    _Float16* __restrict__ WhT, float* __restrict__ s_src,
    float* __restrict__ s_dst, float* __restrict__ blockmax)
{
    __shared__ float Wl[FIN * FOUT];
    __shared__ float al[2 * FOUT];
    __shared__ float red[256];

    const int tid = threadIdx.x;
    for (int i = tid; i < FIN * FOUT; i += 256) Wl[i] = W[i];
    if (tid < 2 * FOUT) al[tid] = attn[tid];
    __syncthreads();

    const int row = blockIdx.x * 256 + tid;   // 32 blocks * 256 = 8192 rows
    float acc[FOUT];
#pragma unroll
    for (int n = 0; n < FOUT; ++n) acc[n] = 0.f;

    const float* hr = h + (size_t)row * FIN;
    for (int k = 0; k < FIN; ++k) {
        const float hv = hr[k];
#pragma unroll
        for (int n = 0; n < FOUT; ++n) acc[n] += hv * Wl[k * FOUT + n];
    }

    float ssrc = 0.f, sdst = 0.f;
#pragma unroll
    for (int n = 0; n < FOUT; ++n) {
        ssrc += acc[n] * al[n];
        sdst += acc[n] * al[FOUT + n];
    }
#pragma unroll
    for (int n = 0; n < FOUT; ++n)
        WhT[(size_t)n * NN + row] = (_Float16)acc[n];   // transposed f16 for WMMA B
    s_src[row] = ssrc;
    s_dst[row] = sdst;

    red[tid] = sdst;
    __syncthreads();
    for (int s = 128; s > 0; s >>= 1) {
        if (tid < s) red[tid] = fmaxf(red[tid], red[tid + s]);
        __syncthreads();
    }
    if (tid == 0) blockmax[blockIdx.x] = red[0];
}

// ---------------- Kernel 2: reduce 32 block maxes -> scalar ----------------
__global__ void gat_maxreduce(const float* __restrict__ blockmax,
                              float* __restrict__ maxdst)
{
    if (threadIdx.x == 0) {
        float m = blockmax[0];
        for (int i = 1; i < 32; ++i) m = fmaxf(m, blockmax[i]);
        *maxdst = m;
    }
}

// ---------------- Kernel 3: fused lrelu+mask+softmax + att@Wh via WMMA ------
// grid.x = NN/16 row blocks; 128 threads = 4 waves; each wave owns K-chunks
// chunk = wave, wave+4, ... (uniform trip count -> EXEC all-ones at every WMMA).
__global__ __launch_bounds__(128) void gat_fused(
    const int* __restrict__ adj, const _Float16* __restrict__ WhT,
    const float* __restrict__ s_src, const float* __restrict__ s_dst,
    const float* __restrict__ maxdst_p, float* __restrict__ out)
{
    __shared__ float accTile[FOUT * FOUT];   // 16x16 (M*16+N)
    __shared__ float lrow[FOUT];             // softmax denominators

    const int tid   = threadIdx.x;
    const int lane  = tid & 31;
    // wave index is uniform across the wave: pin it into an SGPR so the chunk
    // loop compiles to a scalar backedge (s_cmp/s_cbranch), EXEC untouched.
    const int wave  = __builtin_amdgcn_readfirstlane(tid >> 5);  // 0..3
    const int row   = lane & 15;             // M for A-frag / N for B-frag
    const int grp   = lane >> 4;             // 0/1 lane-half
    const int rowBase = blockIdx.x * 16;
    const int gRow  = rowBase + row;

    accTile[tid] = 0.f;
    accTile[tid + 128] = 0.f;
    if (tid < FOUT) lrow[tid] = 0.f;
    __syncthreads();

    const float maxdst = *maxdst_p;
    const float ss  = s_src[gRow];
    const float tM  = ss + maxdst;
    const float M   = tM > 0.f ? tM : ALPHA * tM;   // per-row stable shift
    const float Ml2 = M * LOG2E;                    // pre-scaled for exp2

    v8f   c = {};
    float lsum0 = 0.f, lsum1 = 0.f, lsum2 = 0.f, lsum3 = 0.f;

    const size_t adjRow = (size_t)gRow * NN;
    const int koffA = grp * 8;    // A layout: K {0..7,16..23} vs {8..15,24..31}
    const int koffB = grp * 8;    // B layout: K {0..7,16..23} vs {8..15,24..31}

    for (int chunk = wave; chunk < NN / 32; chunk += 4) {
        const int jb = chunk * 32;

        const int4* ap0 = (const int4*)(adj + adjRow + jb + koffA);
        const int4* ap1 = (const int4*)(adj + adjRow + jb + 16 + koffA);
        int4 a0 = ap0[0], a1 = ap0[1];
        int4 a2 = ap1[0], a3 = ap1[1];

        const float4* sp0 = (const float4*)(s_dst + jb + koffA);
        const float4* sp1 = (const float4*)(s_dst + jb + 16 + koffA);
        float4 d0 = sp0[0], d1 = sp0[1];
        float4 d2 = sp1[0], d3 = sp1[1];

        const int   av[16] = {a0.x,a0.y,a0.z,a0.w, a1.x,a1.y,a1.z,a1.w,
                              a2.x,a2.y,a2.z,a2.w, a3.x,a3.y,a3.z,a3.w};
        const float dv[16] = {d0.x,d0.y,d0.z,d0.w, d1.x,d1.y,d1.z,d1.w,
                              d2.x,d2.y,d2.z,d2.w, d3.x,d3.y,d3.z,d3.w};

        // Phase the computation: 16 independent values per phase so each
        // v_exp_f32 (TRANS) co-issues against its neighbors, no v_nop hazard.
        float ex[16];
#pragma unroll
        for (int i = 0; i < 16; ++i) {
            const float t = ss + dv[i];
            // exp(lrelu(t)-M) = exp2(fma(k*LOG2E, t, -M*LOG2E)), k = t>0?1:ALPHA
            ex[i] = __builtin_fmaf(t > 0.f ? LOG2E : ALPHA * LOG2E, t, -Ml2);
        }
#pragma unroll
        for (int i = 0; i < 16; ++i)
            ex[i] = __builtin_amdgcn_exp2f(ex[i]);

        v16h A;
#pragma unroll
        for (int i = 0; i < 16; ++i) {
            const float p = (av[i] > 0) ? ex[i] : 0.f;   // v_cndmask mask
            A[i] = (_Float16)p;
            ex[i] = p;
        }
        // 4 independent partial-sum chains (dual-issue friendly)
#pragma unroll
        for (int i = 0; i < 4; ++i) {
            lsum0 += ex[i];
            lsum1 += ex[4 + i];
            lsum2 += ex[8 + i];
            lsum3 += ex[12 + i];
        }

        // B fragment: column n=row of Wh, K = jb+koffB .. +7 and jb+16+koffB .. +7
        const v8h* bp = (const v8h*)(WhT + (size_t)row * NN + jb + koffB);
        v8h b0 = bp[0], b1 = bp[2];
        v16h B;
#pragma unroll
        for (int i = 0; i < 8; ++i) { B[i] = b0[i]; B[i + 8] = b1[i]; }

        c = __builtin_amdgcn_wmma_f32_16x16x32_f16(
                false, A, false, B, (short)0, c, false, false);
    }

    float lsum = (lsum0 + lsum1) + (lsum2 + lsum3);
    // combine the two lane-halves of each row's P-sum, then cross-wave via LDS
    lsum += __shfl_xor(lsum, 16, 32);
#pragma unroll
    for (int r = 0; r < 8; ++r)               // C layout: VGPR r -> M = r + grp*8
        atomicAdd(&accTile[(r + koffA) * FOUT + row], c[r]);
    if (grp == 0) atomicAdd(&lrow[row], lsum);
    __syncthreads();

    for (int idx = tid; idx < FOUT * FOUT; idx += 128) {
        const int r = idx >> 4;
        const float v = accTile[idx] / lrow[r];
        // ELU (alpha=1), branch-free
        const float ev = __builtin_amdgcn_exp2f(v * LOG2E) - 1.f;
        out[(size_t)rowBase * FOUT + idx] = v > 0.f ? v : ev;
    }
}

// ---------------- Launcher --------------------------------------------------
extern "C" void kernel_launch(void* const* d_in, const int* in_sizes, int n_in,
                              void* d_out, int out_size, void* d_ws, size_t ws_size,
                              hipStream_t stream) {
    const float* h    = (const float*)d_in[0];   // 8192 x 94
    const int*   adj  = (const int*)d_in[1];     // 8192 x 8192
    const float* W    = (const float*)d_in[2];   // 94 x 16
    const float* attn = (const float*)d_in[3];   // 32 x 1
    float* out = (float*)d_out;                  // 8192 x 16

    char* ws = (char*)d_ws;
    _Float16* WhT   = (_Float16*)ws;                       // 16*8192*2 = 262144 B
    float* s_src    = (float*)(ws + 262144);               // 32768 B
    float* s_dst    = (float*)(ws + 262144 + 32768);       // 32768 B
    float* blockmax = (float*)(ws + 262144 + 65536);       // 128 B
    float* maxdst   = (float*)(ws + 262144 + 65536 + 128); // 4 B

    gat_prep<<<NN / 256, 256, 0, stream>>>(h, W, attn, WhT, s_src, s_dst, blockmax);
    gat_maxreduce<<<1, 32, 0, stream>>>(blockmax, maxdst);
    gat_fused<<<NN / 16, 128, 0, stream>>>(adj, WhT, s_src, s_dst, maxdst, out);
}
